// PointNetPPFC_10771777979128
// MI455X (gfx1250) — matrix operand order repaired and checked by reading, not compiled, not run on hardware
//
#include <hip/hip_runtime.h>
#include <hip/hip_bf16.h>
#include <stdint.h>

// ---------------------------------------------------------------------------
// PointNet++ encoder for MI455X (gfx1250).
// GEMMs (99% of FLOPs) run on v_wmma_f32_16x16x32_bf16 with f32 accumulation.
// GEMM k-loop is software-pipelined: next k-step's A/B fragments are loaded
// into distinct registers before the current step's WMMAs, so the memory
// latency overlaps matrix math instead of stalling on s_wait_loadcnt 0.
// ---------------------------------------------------------------------------

#define BATCH   16
#define NPTS    2048
#define KNN     10
#define EDGES   (BATCH * NPTS * KNN)      // 327680 rows, multiple of 128
#define ZDIM    128
#define BN_EPS  1e-5f

typedef __attribute__((ext_vector_type(16))) __bf16 v16bf;
typedef __attribute__((ext_vector_type(8)))  float  v8f;

union Frag { uint4 u[2]; v16bf v; };

__device__ __forceinline__ unsigned short f2bf(float x) {
    unsigned int u = __float_as_uint(x);
    unsigned int r = u + 0x7FFFu + ((u >> 16) & 1u);   // round-to-nearest-even
    return (unsigned short)(r >> 16);
}

// ---------------------------------------------------------------------------
// 1) kNN: per-batch positions staged in LDS (2048*2 f32 = 16KB), each thread
//    scans all N points keeping a sorted top-10 (strict < => stable like
//    jax.lax.top_k on -d).
// ---------------------------------------------------------------------------
__global__ void knn_kernel(const float* __restrict__ pos, int* __restrict__ idx) {
    __shared__ float sx[NPTS], sy[NPTS];
    const int blocksPerBatch = NPTS / 256;
    const int b  = blockIdx.x / blocksPerBatch;
    const int q0 = (blockIdx.x % blocksPerBatch) * 256;
    const float* pb = pos + (size_t)b * NPTS * 2;
    for (int i = threadIdx.x; i < NPTS; i += 256) { sx[i] = pb[2*i]; sy[i] = pb[2*i+1]; }
    __syncthreads();
    const int q  = q0 + threadIdx.x;
    const float qx = sx[q], qy = sy[q];
    float bd[KNN]; int bi[KNN];
#pragma unroll
    for (int p = 0; p < KNN; ++p) { bd[p] = 3.4e38f; bi[p] = 0; }
    for (int j = 0; j < NPTS; ++j) {
        float dx = sx[j] - qx, dy = sy[j] - qy;
        float d  = dx*dx + dy*dy;
        if (d < bd[KNN-1]) {
            bd[KNN-1] = d; bi[KNN-1] = j;
#pragma unroll
            for (int p = KNN-1; p > 0; --p) {
                if (bd[p] < bd[p-1]) {
                    float td = bd[p]; bd[p] = bd[p-1]; bd[p-1] = td;
                    int   ti = bi[p]; bi[p] = bi[p-1]; bi[p-1] = ti;
                }
            }
        }
    }
    int* out = idx + ((size_t)b * NPTS + q) * KNN;
#pragma unroll
    for (int p = 0; p < KNN; ++p) out[p] = bi[p];
}

// ---------------------------------------------------------------------------
// 2) fp32 features -> bf16 feature bank (conv1 input)
// ---------------------------------------------------------------------------
__global__ void feat2bf_kernel(const float* __restrict__ f, unsigned short* __restrict__ h, int n) {
    int i = blockIdx.x * blockDim.x + threadIdx.x;
    if (i < n) h[i] = f2bf(f[i]);
}

// ---------------------------------------------------------------------------
// 3) Build edge matrix A [E, Kpad] (bf16): concat(gather(h), pos_j - pos_i),
//    zero-padded in K up to a multiple of 32.
// ---------------------------------------------------------------------------
__global__ void build_a_kernel(const unsigned short* __restrict__ Hbf,
                               const float* __restrict__ pos,
                               const int* __restrict__ idx,
                               unsigned short* __restrict__ A,
                               int Cin, int Kpad, long long total) {
    long long i = (long long)blockIdx.x * blockDim.x + threadIdx.x;
    if (i >= total) return;
    int c        = (int)(i % Kpad);
    long long e  = i / Kpad;            // edge row
    long long bn = e / KNN;             // b*N + n
    unsigned short v = 0;
    int j = idx[e];
    long long bbase = (bn / NPTS) * NPTS;
    if (c < Cin) {
        v = Hbf[(bbase + j) * Cin + c];
    } else if (c < Cin + 2) {
        const float* pb = pos + bbase * 2;
        int n  = (int)(bn % NPTS);
        int ax = c - Cin;
        v = f2bf(pb[2*j + ax] - pb[2*n + ax]);
    }
    A[i] = v;
}

// ---------------------------------------------------------------------------
// 4) Pack weights: W fp32 [cinA, Cout] (row-major)  ->  WT bf16 [Cout, Kpad]
//    (transposed + K-padded with zeros) so B-fragments are contiguous b128s.
// ---------------------------------------------------------------------------
__global__ void pack_w_kernel(const float* __restrict__ W, unsigned short* __restrict__ WT,
                              int cinA, int Kpad, int Cout) {
    int i = blockIdx.x * blockDim.x + threadIdx.x;
    if (i >= Cout * Kpad) return;
    int k = i % Kpad, n = i / Kpad;
    WT[i] = (k < cinA) ? f2bf(W[(size_t)k * Cout + n]) : (unsigned short)0;
}

// ---------------------------------------------------------------------------
// 5) WMMA GEMM:  C[M,Cout] = A[M,Kpad](bf16) x WT[Cout,Kpad](bf16)^T + bias
//    256 threads = 8 waves; each wave -> one 16x64 output strip.
//    Fragment layouts (contiguous k order, not the ISA table order; A and B use the same):
//      A 16x32: lane m = L%16, 16 contiguous K at (L/16)*16 (two b128s)
//      B 32x16: lane n = L%16, 16 contiguous K at (L/16)*16 (two b128s)
//    Software-pipelined: prefetch next k-step fragments before current WMMAs.
// ---------------------------------------------------------------------------
#define WMMA_BF16(AF, BF, ACC) \
    __builtin_amdgcn_wmma_f32_16x16x32_bf16(false, (AF).v, false, (BF).v, (short)0, (ACC), false, false)

__global__ void __launch_bounds__(256)
gemm_bf16_wmma(const unsigned short* __restrict__ A,
               const unsigned short* __restrict__ WT,
               const float* __restrict__ bias,
               float* __restrict__ C,
               int M, int Kpad, int Cout) {
    const int lane = threadIdx.x & 31;
    const int wave = threadIdx.x >> 5;
    const int tileM = (blockIdx.x * 8 + wave) * 16;
    const int tileN = blockIdx.y * 64;
    if (tileM >= M) return;

    v8f z = {0.f,0.f,0.f,0.f,0.f,0.f,0.f,0.f};
    v8f acc0 = z, acc1 = z, acc2 = z, acc3 = z;

    const int half = lane >> 4;                 // 0: lanes 0-15, 1: lanes 16-31
    const unsigned short* Arow = A + (size_t)(tileM + (lane & 15)) * Kpad + half * 16;
    const unsigned short* B0 = WT + (size_t)(tileN +  0 + (lane & 15)) * Kpad + half * 16;
    const unsigned short* B1 = WT + (size_t)(tileN + 16 + (lane & 15)) * Kpad + half * 16;
    const unsigned short* B2 = WT + (size_t)(tileN + 32 + (lane & 15)) * Kpad + half * 16;
    const unsigned short* B3 = WT + (size_t)(tileN + 48 + (lane & 15)) * Kpad + half * 16;

#define LOAD_A(F, KK)  { (F).u[0] = *(const uint4*)(Arow + (KK));       \
                         (F).u[1] = *(const uint4*)(Arow + (KK) + 8); }
#define LOAD_B(F, BP, KK) { (F).u[0] = *(const uint4*)((BP) + (KK));      \
                            (F).u[1] = *(const uint4*)((BP) + (KK) + 8); }

    Frag af, bf0, bf1, bf2, bf3;
    LOAD_A(af, 0);
    LOAD_B(bf0, B0, 0); LOAD_B(bf1, B1, 0);
    LOAD_B(bf2, B2, 0); LOAD_B(bf3, B3, 0);

    for (int kk = 0; kk < Kpad; kk += 32) {
        // Prefetch next k-step into distinct registers (clamped on last iter).
        const int kn = (kk + 32 < Kpad) ? (kk + 32) : kk;
        Frag an, bn0, bn1, bn2, bn3;
        LOAD_A(an, kn);
        LOAD_B(bn0, B0, kn); LOAD_B(bn1, B1, kn);
        LOAD_B(bn2, B2, kn); LOAD_B(bn3, B3, kn);

        // Matrix math on already-resident fragments (no load dependence).
        acc0 = WMMA_BF16(af, bf0, acc0);
        acc1 = WMMA_BF16(af, bf1, acc1);
        acc2 = WMMA_BF16(af, bf2, acc2);
        acc3 = WMMA_BF16(af, bf3, acc3);

        af = an; bf0 = bn0; bf1 = bn1; bf2 = bn2; bf3 = bn3;
    }
#undef LOAD_A
#undef LOAD_B

    v8f accs[4] = {acc0, acc1, acc2, acc3};
#pragma unroll
    for (int j = 0; j < 4; ++j) {
        int col = tileN + j * 16 + (lane & 15);
        float bv = bias[col];
#pragma unroll
        for (int r = 0; r < 8; ++r) {           // C/D: VGPR r -> row r (+8 for half 1)
            int row = tileM + r + (half << 3);
            C[(size_t)row * Cout + col] = accs[j][r] + bv;
        }
    }
}

// ---------------------------------------------------------------------------
// 6) BatchNorm statistics: per-channel sum / sumsq over E rows (f32 atomics).
// ---------------------------------------------------------------------------
__global__ void zero_kernel(float* __restrict__ p, int n) {
    int i = blockIdx.x * blockDim.x + threadIdx.x;
    if (i < n) p[i] = 0.f;
}

__global__ void bn_stats_kernel(const float* __restrict__ C, float* __restrict__ stats,
                                int E, int Cout) {
    int c = blockIdx.y * 64 + threadIdx.x;      // blockDim.x == 64
    float s = 0.f, s2 = 0.f;
    for (int r = blockIdx.x; r < E; r += gridDim.x) {
        float x = C[(size_t)r * Cout + c];
        s += x; s2 += x * x;
    }
    atomicAdd(&stats[c], s);
    atomicAdd(&stats[Cout + c], s2);
}

// ---------------------------------------------------------------------------
// 7) BN + ReLU -> bf16 (input to GEMM2)
// ---------------------------------------------------------------------------
__global__ void bn_relu_bf16_kernel(const float* __restrict__ C, const float* __restrict__ stats,
                                    const float* __restrict__ gamma, const float* __restrict__ beta,
                                    unsigned short* __restrict__ A,
                                    int Cout, long long total, float invE) {
    long long i = (long long)blockIdx.x * blockDim.x + threadIdx.x;
    if (i >= total) return;
    int c = (int)(i % Cout);
    float mean = stats[c] * invE;
    float var  = stats[Cout + c] * invE - mean * mean;
    float y = (C[i] - mean) * rsqrtf(var + BN_EPS) * gamma[c] + beta[c];
    A[i] = f2bf(fmaxf(y, 0.f));
}

// ---------------------------------------------------------------------------
// 8) BN2 + max over K neighbors + ReLU -> H (bf16 for next gather, f32 pool)
// ---------------------------------------------------------------------------
__global__ void bn_max_relu_kernel(const float* __restrict__ C, const float* __restrict__ stats,
                                   const float* __restrict__ gamma, const float* __restrict__ beta,
                                   unsigned short* __restrict__ Hbf, float* __restrict__ Hf,
                                   int Cout, float invE) {
    int i = blockIdx.x * blockDim.x + threadIdx.x;
    if (i >= BATCH * NPTS * Cout) return;
    int c  = i % Cout;
    int bn = i / Cout;
    float mean = stats[c] * invE;
    float inv  = rsqrtf(stats[Cout + c] * invE - mean * mean + BN_EPS);
    float ga = gamma[c], be = beta[c];
    float m = -3.4e38f;
    size_t base = (size_t)bn * KNN * Cout + c;
    for (int k = 0; k < KNN; ++k) {
        float x = C[base + (size_t)k * Cout];
        m = fmaxf(m, (x - mean) * inv * ga + be);
    }
    float h = fmaxf(m, 0.f);
    Hbf[i] = f2bf(h);
    Hf[i]  = h;
}

// ---------------------------------------------------------------------------
// 9) Global max pool over N -> g [B, 512]
// ---------------------------------------------------------------------------
__global__ void gmax_kernel(const float* __restrict__ Hf, float* __restrict__ g) {
    int i = blockIdx.x * blockDim.x + threadIdx.x;
    if (i >= BATCH * 512) return;
    int c = i % 512, b = i / 512;
    float m = -3.4e38f;
    const float* base = Hf + (size_t)b * NPTS * 512 + c;
    for (int n = 0; n < NPTS; ++n) m = fmaxf(m, base[(size_t)n * 512]);
    g[i] = m;
}

// ---------------------------------------------------------------------------
// 10) SiLU FC head: 16x512 -> 256 -> 128 -> 128 (single block)
// ---------------------------------------------------------------------------
__device__ __forceinline__ float silu(float x) { return x / (1.f + expf(-x)); }

__global__ void __launch_bounds__(256)
fc_head_kernel(const float* __restrict__ g,
               const float* __restrict__ W1, const float* __restrict__ b1,
               const float* __restrict__ W2, const float* __restrict__ b2,
               const float* __restrict__ W3, const float* __restrict__ b3,
               float* __restrict__ out) {
    __shared__ float h1[BATCH * 256];
    __shared__ float h2[BATCH * 128];
    int tid = threadIdx.x;
    for (int o = tid; o < BATCH * 256; o += 256) {
        int b = o / 256, c = o % 256;
        float s = b1[c];
        for (int i = 0; i < 512; ++i) s += g[b * 512 + i] * W1[i * 256 + c];
        h1[o] = silu(s);
    }
    __syncthreads();
    for (int o = tid; o < BATCH * 128; o += 256) {
        int b = o / 128, c = o % 128;
        float s = b2[c];
        for (int i = 0; i < 256; ++i) s += h1[b * 256 + i] * W2[i * 128 + c];
        h2[o] = silu(s);
    }
    __syncthreads();
    for (int o = tid; o < BATCH * ZDIM; o += 256) {
        int b = o / ZDIM, c = o % ZDIM;
        float s = b3[c];
        for (int i = 0; i < 128; ++i) s += h2[b * 128 + i] * W3[i * ZDIM + c];
        out[o] = s;
    }
}

// ---------------------------------------------------------------------------
// Host-side orchestration.
// Input order (JAX pytree alphabetical flatten):
//   0: features [B,N,2]
//   per conv layer L (base = 1 + 8*L): bn1.beta, bn1.gamma, bn2.beta, bn2.gamma,
//                                      lin1.W, lin1.b, lin2.W, lin2.b
//   33/34: fc1.W/b   35/36: fc2.W/b   37/38: fc3.W/b
//   39: position [B,N,2]
// ---------------------------------------------------------------------------
extern "C" void kernel_launch(void* const* d_in, const int* in_sizes, int n_in,
                              void* d_out, int out_size, void* d_ws, size_t ws_size,
                              hipStream_t stream) {
    (void)in_sizes; (void)n_in; (void)out_size; (void)ws_size;

    const float* features = (const float*)d_in[0];
    const float* position = (const float*)d_in[39];

    // Workspace layout (aligned to 256B); total ~1.11 GB
    char* ws = (char*)d_ws;
    size_t off = 0;
    auto alloc = [&](size_t bytes) { void* p = ws + off; off = (off + bytes + 255) & ~(size_t)255; return p; };
    int*            idx   = (int*)           alloc((size_t)EDGES * 4);                 // 1.3 MB
    unsigned short* Abuf  = (unsigned short*)alloc((size_t)EDGES * 512 * 2);           // 336 MB
    float*          Cbuf  = (float*)         alloc((size_t)EDGES * 512 * 4);           // 671 MB
    unsigned short* Hbf   = (unsigned short*)alloc((size_t)BATCH * NPTS * 512 * 2);    // 34 MB
    float*          Hf    = (float*)         alloc((size_t)BATCH * NPTS * 512 * 4);    // 67 MB
    unsigned short* WTbuf = (unsigned short*)alloc((size_t)512 * 512 * 2);             // 0.5 MB
    float*          stats = (float*)         alloc(1024 * 4);
    float*          gpool = (float*)         alloc((size_t)BATCH * 512 * 4);

    // 1) kNN
    knn_kernel<<<BATCH * (NPTS / 256), 256, 0, stream>>>(position, idx);

    // 2) features -> bf16 bank
    {
        int n = BATCH * NPTS * 2;
        feat2bf_kernel<<<(n + 255) / 256, 256, 0, stream>>>(features, Hbf, n);
    }

    struct LayerCfg { int cin, kpad1, cout, pbase; };
    const LayerCfg layers[4] = {
        {   2,  32,  64,  1 },
        {  64,  96, 128,  9 },
        { 128, 160, 256, 17 },
        { 256, 288, 512, 25 },
    };

    const float invE = 1.0f / (float)EDGES;

    for (int L = 0; L < 4; ++L) {
        const LayerCfg& lc = layers[L];
        const float* bn1_beta  = (const float*)d_in[lc.pbase + 0];
        const float* bn1_gamma = (const float*)d_in[lc.pbase + 1];
        const float* bn2_beta  = (const float*)d_in[lc.pbase + 2];
        const float* bn2_gamma = (const float*)d_in[lc.pbase + 3];
        const float* W1        = (const float*)d_in[lc.pbase + 4];
        const float* b1        = (const float*)d_in[lc.pbase + 5];
        const float* W2        = (const float*)d_in[lc.pbase + 6];
        const float* b2        = (const float*)d_in[lc.pbase + 7];

        // Build A [E, Kpad1]
        {
            long long total = (long long)EDGES * lc.kpad1;
            int blocks = (int)((total + 255) / 256);
            build_a_kernel<<<blocks, 256, 0, stream>>>(Hbf, position, idx, Abuf,
                                                       lc.cin, lc.kpad1, total);
        }
        // Pack W1^T and GEMM1 -> Cbuf [E, Cout]
        {
            int n = lc.cout * lc.kpad1;
            pack_w_kernel<<<(n + 255) / 256, 256, 0, stream>>>(W1, WTbuf, lc.cin + 2, lc.kpad1, lc.cout);
            dim3 grid(EDGES / 128, lc.cout / 64);
            gemm_bf16_wmma<<<grid, 256, 0, stream>>>(Abuf, WTbuf, b1, Cbuf, EDGES, lc.kpad1, lc.cout);
        }
        // BN1 stats + BN1+ReLU -> A2 (bf16, reuse Abuf, [E, Cout])
        {
            zero_kernel<<<4, 256, 0, stream>>>(stats, 2 * lc.cout);
            dim3 sgrid(1024, lc.cout / 64);
            bn_stats_kernel<<<sgrid, 64, 0, stream>>>(Cbuf, stats, EDGES, lc.cout);
            long long total = (long long)EDGES * lc.cout;
            bn_relu_bf16_kernel<<<(int)((total + 255) / 256), 256, 0, stream>>>(
                Cbuf, stats, bn1_gamma, bn1_beta, Abuf, lc.cout, total, invE);
        }
        // Pack W2^T and GEMM2 -> Cbuf [E, Cout]
        {
            int n = lc.cout * lc.cout;
            pack_w_kernel<<<(n + 255) / 256, 256, 0, stream>>>(W2, WTbuf, lc.cout, lc.cout, lc.cout);
            dim3 grid(EDGES / 128, lc.cout / 64);
            gemm_bf16_wmma<<<grid, 256, 0, stream>>>(Abuf, WTbuf, b2, Cbuf, EDGES, lc.cout, lc.cout);
        }
        // BN2 stats + BN2 + max-over-K + ReLU -> H
        {
            zero_kernel<<<4, 256, 0, stream>>>(stats, 2 * lc.cout);
            dim3 sgrid(1024, lc.cout / 64);
            bn_stats_kernel<<<sgrid, 64, 0, stream>>>(Cbuf, stats, EDGES, lc.cout);
            int n = BATCH * NPTS * lc.cout;
            bn_max_relu_kernel<<<(n + 255) / 256, 256, 0, stream>>>(
                Cbuf, stats, bn2_gamma, bn2_beta, Hbf, Hf, lc.cout, invE);
        }
    }

    // Global max pool + FC head
    {
        int n = BATCH * 512;
        gmax_kernel<<<(n + 255) / 256, 256, 0, stream>>>(Hf, gpool);
        fc_head_kernel<<<1, 256, 0, stream>>>(gpool,
            (const float*)d_in[33], (const float*)d_in[34],
            (const float*)d_in[35], (const float*)d_in[36],
            (const float*)d_in[37], (const float*)d_in[38],
            (float*)d_out);
    }
}
